// LlamaDecoderLayer_33200097198615
// MI455X (gfx1250) — compile-verified, matrix operand order checked by reading, not run on hardware
//
#include <hip/hip_runtime.h>
#include <hip/hip_bf16.h>

// ---- problem constants (match reference) ----
#define TT   2048
#define HH   4096
#define HQn  32
#define HKVn 8
#define DD   128
#define II   14336
#define THETA 500000.0f
#define EPS_  1e-5f

typedef __attribute__((ext_vector_type(8)))  float   v8f;
typedef __attribute__((ext_vector_type(8)))  __bf16  v8bf;
typedef __attribute__((ext_vector_type(16))) __bf16  v16bf;
typedef __attribute__((ext_vector_type(4)))  int     v4i;

#if defined(__AMDGCN__) && __has_builtin(__builtin_amdgcn_global_load_async_to_lds_b128) && \
    __has_builtin(__builtin_amdgcn_s_wait_asynccnt)
#define USE_ASYNC_LDS 1
#else
#define USE_ASYNC_LDS 0
#endif

static __device__ __forceinline__ v16bf bf_combine(v8bf lo, v8bf hi) {
    return __builtin_shufflevector(lo, hi, 0,1,2,3,4,5,6,7,8,9,10,11,12,13,14,15);
}

// async b128 copy global -> LDS (CDNA5 GLOBAL_LOAD_ASYNC_TO_LDS_B128), with
// a register-path fallback when the builtin is unavailable (e.g. host pass).
static __device__ __forceinline__ void copy16_g2lds(const void* gsrc, void* ldst) {
#if USE_ASYNC_LDS
    __builtin_amdgcn_global_load_async_to_lds_b128(
        (__attribute__((address_space(1))) v4i*)(gsrc),
        (__attribute__((address_space(3))) v4i*)(ldst), 0, 0);
#else
    *(v8bf*)ldst = *(const v8bf*)gsrc;
#endif
}

static __device__ __forceinline__ void wait_async_zero() {
#if USE_ASYNC_LDS
    __builtin_amdgcn_s_wait_asynccnt(0);
#endif
}

// =====================================================================
// RMSNorm: one block per row; fp32 in -> bf16 out
// =====================================================================
__global__ __launch_bounds__(256)
void rmsnorm_bf16_kernel(const float* __restrict__ x, const float* __restrict__ w,
                         __bf16* __restrict__ out) {
    __shared__ float red[256];
    const int t = blockIdx.x;
    const float* xr = x + (size_t)t * HH;
    float ss = 0.f;
    for (int i = threadIdx.x; i < HH; i += 256) { float v = xr[i]; ss += v * v; }
    red[threadIdx.x] = ss;
    __syncthreads();
    for (int s = 128; s > 0; s >>= 1) {
        if (threadIdx.x < s) red[threadIdx.x] += red[threadIdx.x + s];
        __syncthreads();
    }
    const float scale = rsqrtf(red[0] * (1.0f / HH) + EPS_);
    __bf16* orow = out + (size_t)t * HH;
    for (int i = threadIdx.x; i < HH; i += 256)
        orow[i] = (__bf16)(xr[i] * scale * w[i]);
}

// =====================================================================
// RoPE: fp32 (T, nh, 128) -> bf16 with rotation
// =====================================================================
__global__ __launch_bounds__(256)
void rope_bf16_kernel(const float* __restrict__ src, const int* __restrict__ pos,
                      __bf16* __restrict__ dst, int nh) {
    int idx = blockIdx.x * 256 + threadIdx.x;
    int i = idx & 63;
    int h = (idx >> 6) % nh;
    int t = idx / (64 * nh);
    if (t >= TT) return;
    float inv_freq = __powf(THETA, -(2.0f * (float)i) / (float)DD);
    float fr = (float)pos[t] * inv_freq;
    float c = __cosf(fr), s = __sinf(fr);
    size_t base = ((size_t)t * nh + h) * DD;
    float x1 = src[base + i];
    float x2 = src[base + i + 64];
    dst[base + i]      = (__bf16)(x1 * c - x2 * s);
    dst[base + i + 64] = (__bf16)(x2 * c + x1 * s);
}

__global__ __launch_bounds__(256)
void cvt_f32_bf16_kernel(const float* __restrict__ src, __bf16* __restrict__ dst, int n) {
    int i = blockIdx.x * 256 + threadIdx.x;
    if (i < n) dst[i] = (__bf16)src[i];
}

// =====================================================================
// Double-buffered WMMA GEMM:  C[M,N] = A_bf16[M,K] @ W_f32[N,K]^T
// 128x128 block tile, BK=64, 8 waves (4x2), async A staging.
// EPI: 0 = store f32 C ; 1 = store f32 (C+resid) ; 2 = bf16(silu(gate)*C)
// =====================================================================
template<int EPI>
__global__ __launch_bounds__(256)
void gemm_bf16_wmma_kernel(const __bf16* __restrict__ A, const float* __restrict__ W,
                           float* __restrict__ outf, __bf16* __restrict__ outb,
                           const float* __restrict__ resid, const float* __restrict__ gatef,
                           int N, int K) {
    __shared__ __bf16 lA[2][128 * 72];   // [m][k], stride 72 (16B-aligned rows)
    __shared__ __bf16 lB[2][128 * 72];   // [n][k], stride 72

    const int tid    = threadIdx.x;
    const int lane   = tid & 31;
    const int wid    = tid >> 5;
    const int wave_m = wid & 3;   // 4 waves along M, 32 rows each
    const int wave_n = wid >> 2;  // 2 waves along N, 64 cols each
    const int m0 = blockIdx.y * 128;
    const int n0 = blockIdx.x * 128;
    const int khalf = lane >> 4;
    const int l15   = lane & 15;

    v8f acc[2][4];
    #pragma unroll
    for (int mt = 0; mt < 2; ++mt)
        #pragma unroll
        for (int nt = 0; nt < 4; ++nt)
            acc[mt][nt] = (v8f)(0.0f);

    auto stage = [&](int buf, int k0) {
        // ---- A tile: 128x64 bf16, 1024 16B chunks, async direct to LDS ----
        #pragma unroll
        for (int i = 0; i < 4; ++i) {
            int c   = tid + i * 256;
            int row = c >> 3;
            int cg  = (c & 7) << 3;
            copy16_g2lds(A + (size_t)(m0 + row) * K + k0 + cg,
                         &lA[buf][row * 72 + cg]);
        }
        // ---- W tile: 128x64 f32 -> bf16 through VGPRs ----
        #pragma unroll
        for (int i = 0; i < 4; ++i) {
            int c   = tid + i * 256;
            int row = c >> 3;
            int cg  = (c & 7) << 3;
            const float4* p = (const float4*)(W + (size_t)(n0 + row) * K + k0 + cg);
            float tmp[8];
            *(float4*)(tmp)     = p[0];
            *(float4*)(tmp + 4) = p[1];
            v8bf v;
            #pragma unroll
            for (int e = 0; e < 8; ++e) v[e] = (__bf16)tmp[e];
            *(v8bf*)(&lB[buf][row * 72 + cg]) = v;
        }
    };

    auto compute = [&](int buf) {
        #pragma unroll
        for (int kc = 0; kc < 64; kc += 32) {
            v16bf afrag[2];
            #pragma unroll
            for (int mt = 0; mt < 2; ++mt) {
                int row = wave_m * 32 + mt * 16 + l15;
                v8bf lo = *(const v8bf*)(&lA[buf][row * 72 + kc + khalf * 8]);
                v8bf hi = *(const v8bf*)(&lA[buf][row * 72 + kc + 16 + khalf * 8]);
                afrag[mt] = bf_combine(lo, hi);
            }
            #pragma unroll
            for (int nt = 0; nt < 4; ++nt) {
                int col = wave_n * 64 + nt * 16 + l15;
                v8bf lo = *(const v8bf*)(&lB[buf][col * 72 + kc + khalf * 8]);
                v8bf hi = *(const v8bf*)(&lB[buf][col * 72 + kc + 16 + khalf * 8]);
                v16bf bfrag = bf_combine(lo, hi);
                #pragma unroll
                for (int mt = 0; mt < 2; ++mt)
                    acc[mt][nt] = __builtin_amdgcn_wmma_f32_16x16x32_bf16(
                        false, afrag[mt], false, bfrag, (short)0, acc[mt][nt], false, false);
            }
        }
    };

    const int nk = K >> 6;
    stage(0, 0);
    wait_async_zero();
    __syncthreads();
    for (int kt = 0; kt < nk; ++kt) {
        if (kt + 1 < nk) stage((kt + 1) & 1, (kt + 1) << 6);
        compute(kt & 1);
        wait_async_zero();
        __syncthreads();
    }

    // ---- epilogue ----
    #pragma unroll
    for (int mt = 0; mt < 2; ++mt) {
        int mb = m0 + wave_m * 32 + mt * 16 + (khalf << 3);
        #pragma unroll
        for (int nt = 0; nt < 4; ++nt) {
            int gc = n0 + wave_n * 64 + nt * 16 + l15;
            #pragma unroll
            for (int r = 0; r < 8; ++r) {
                size_t idx = (size_t)(mb + r) * N + gc;
                float v = acc[mt][nt][r];
                if (EPI == 0) {
                    outf[idx] = v;
                } else if (EPI == 1) {
                    outf[idx] = v + resid[idx];
                } else {
                    float g = gatef[idx];
                    float sg = g / (1.0f + __expf(-g));   // silu(gate)
                    outb[idx] = (__bf16)(sg * v);
                }
            }
        }
    }
}

// =====================================================================
// Flash attention (causal, GQA 4:1).  grid = (T/64, HQ), 128 threads.
// =====================================================================
__global__ __launch_bounds__(128)
void flash_attn_kernel(const __bf16* __restrict__ Q, const __bf16* __restrict__ Kb,
                       const __bf16* __restrict__ Vb, __bf16* __restrict__ O) {
    __shared__ __bf16 Klds[64 * 136];       // [n][d]   stride 136
    __shared__ __bf16 Vt[128 * 72];         // [d][n]   stride 72  (V transposed)
    __shared__ __bf16 Plds[4 * 16 * 72];    // per-wave [m][n] stride 72

    const int tid  = threadIdx.x;
    const int lane = tid & 31;
    const int wid  = tid >> 5;
    const int l15  = lane & 15;
    const int khalf = lane >> 4;
    const int head  = blockIdx.y;
    const int kvh   = head >> 2;            // GQA group of 4
    const int qtile = blockIdx.x;
    const int qrow0 = qtile * 64 + wid * 16;
    const float scale = 0.08838834764831845f;   // 1/sqrt(128)

    v16bf qfrag[4];
    {
        size_t qbase = (size_t)(qrow0 + l15) * (HQn * DD) + (size_t)head * DD;
        #pragma unroll
        for (int c = 0; c < 4; ++c) {
            int d0 = c * 32;
            v8bf lo = *(const v8bf*)(Q + qbase + d0 + khalf * 8);
            v8bf hi = *(const v8bf*)(Q + qbase + d0 + 16 + khalf * 8);
            qfrag[c] = bf_combine(lo, hi);
        }
    }

    v8f   accO[8];
    float mrow[8], lrow[8];
    #pragma unroll
    for (int dt = 0; dt < 8; ++dt) accO[dt] = (v8f)(0.0f);
    #pragma unroll
    for (int r = 0; r < 8; ++r) { mrow[r] = -3.0e38f; lrow[r] = 0.0f; }

    __bf16* Pw = &Plds[wid * 16 * 72];

    for (int j = 0; j <= qtile; ++j) {
        // ---- K tile async -> LDS ; V tile transposed through regs ----
        #pragma unroll
        for (int i = 0; i < 8; ++i) {
            int c  = tid + i * 128;
            int n  = c >> 4;
            int dg = (c & 15) << 3;
            size_t src = (size_t)(j * 64 + n) * (HKVn * DD) + (size_t)kvh * DD + dg;
            copy16_g2lds(Kb + src, &Klds[n * 136 + dg]);
            v8bf vv = *(const v8bf*)(Vb + src);
            #pragma unroll
            for (int e = 0; e < 8; ++e) Vt[(dg + e) * 72 + n] = vv[e];
        }
        wait_async_zero();
        __syncthreads();

        // ---- S = Q K^T ----
        v8f s[4];
        #pragma unroll
        for (int nt = 0; nt < 4; ++nt) {
            s[nt] = (v8f)(0.0f);
            int n = nt * 16 + l15;
            #pragma unroll
            for (int c = 0; c < 4; ++c) {
                int d0 = c * 32;
                v8bf lo = *(const v8bf*)(&Klds[n * 136 + d0 + khalf * 8]);
                v8bf hi = *(const v8bf*)(&Klds[n * 136 + d0 + 16 + khalf * 8]);
                s[nt] = __builtin_amdgcn_wmma_f32_16x16x32_bf16(
                    false, qfrag[c], false, bf_combine(lo, hi), (short)0, s[nt], false, false);
            }
        }

        // ---- scale + causal mask ----
        #pragma unroll
        for (int nt = 0; nt < 4; ++nt) {
            int col = j * 64 + nt * 16 + l15;
            #pragma unroll
            for (int r = 0; r < 8; ++r) {
                int row = qrow0 + (khalf << 3) + r;
                float v = s[nt][r] * scale;
                s[nt][r] = (col > row) ? -3.0e38f : v;
            }
        }

        // ---- online softmax ----
        #pragma unroll
        for (int r = 0; r < 8; ++r) {
            float mt = fmaxf(fmaxf(s[0][r], s[1][r]), fmaxf(s[2][r], s[3][r]));
            #pragma unroll
            for (int off = 8; off >= 1; off >>= 1)
                mt = fmaxf(mt, __shfl_xor(mt, off, 32));
            float newm = fmaxf(mrow[r], mt);
            float fac  = __expf(mrow[r] - newm);
            mrow[r] = newm;
            lrow[r] *= fac;
            #pragma unroll
            for (int dt = 0; dt < 8; ++dt) accO[dt][r] *= fac;
            float rs = 0.f;
            #pragma unroll
            for (int nt = 0; nt < 4; ++nt) {
                float p = __expf(s[nt][r] - newm);
                s[nt][r] = p;
                rs += p;
            }
            #pragma unroll
            for (int off = 8; off >= 1; off >>= 1)
                rs += __shfl_xor(rs, off, 32);
            lrow[r] += rs;
        }

        // ---- stage P (bf16) into per-wave LDS ----
        #pragma unroll
        for (int nt = 0; nt < 4; ++nt) {
            int col = nt * 16 + l15;
            #pragma unroll
            for (int r = 0; r < 8; ++r)
                Pw[((khalf << 3) + r) * 72 + col] = (__bf16)s[nt][r];
        }

        // ---- O += P V ----
        #pragma unroll
        for (int h2 = 0; h2 < 2; ++h2) {
            int nb = h2 * 32;
            v8bf plo = *(const v8bf*)(&Pw[l15 * 72 + nb + khalf * 8]);
            v8bf phi = *(const v8bf*)(&Pw[l15 * 72 + nb + 16 + khalf * 8]);
            v16bf pfrag = bf_combine(plo, phi);
            #pragma unroll
            for (int dt = 0; dt < 8; ++dt) {
                int d = dt * 16 + l15;
                v8bf lo = *(const v8bf*)(&Vt[d * 72 + nb + khalf * 8]);
                v8bf hi = *(const v8bf*)(&Vt[d * 72 + nb + 16 + khalf * 8]);
                accO[dt] = __builtin_amdgcn_wmma_f32_16x16x32_bf16(
                    false, pfrag, false, bf_combine(lo, hi), (short)0, accO[dt], false, false);
            }
        }
        __syncthreads();
    }

    // ---- normalize + store bf16 (T, HQ*D) ----
    #pragma unroll
    for (int r = 0; r < 8; ++r) {
        float inv = 1.0f / lrow[r];
        int row = qrow0 + (khalf << 3) + r;
        #pragma unroll
        for (int dt = 0; dt < 8; ++dt)
            O[(size_t)row * (HQn * DD) + (size_t)head * DD + dt * 16 + l15] =
                (__bf16)(accO[dt][r] * inv);
    }
}

// =====================================================================
// host-side launch
// =====================================================================
extern "C" void kernel_launch(void* const* d_in, const int* in_sizes, int n_in,
                              void* d_out, int out_size, void* d_ws, size_t ws_size,
                              hipStream_t stream) {
    const float* hidden = (const float*)d_in[0];
    const int*   pos    = (const int*)  d_in[1];
    const float* w_q    = (const float*)d_in[2];
    const float* w_k    = (const float*)d_in[3];
    const float* w_v    = (const float*)d_in[4];
    const float* w_o    = (const float*)d_in[5];
    const float* w_gate = (const float*)d_in[6];
    const float* w_up   = (const float*)d_in[7];
    const float* w_down = (const float*)d_in[8];
    const float* rms1_w = (const float*)d_in[9];
    const float* rms2_w = (const float*)d_in[10];
    float* out = (float*)d_out;

    char* ws = (char*)d_ws;
    size_t off = 0;
    auto take = [&](size_t bytes) { char* p = ws + off; off += (bytes + 255) & ~(size_t)255; return p; };

    float*  qf32  = (float*) take((size_t)TT * HQn  * DD * 4);
    float*  kf32  = (float*) take((size_t)TT * HKVn * DD * 4);
    float*  vf32  = (float*) take((size_t)TT * HKVn * DD * 4);
    float*  gatef = (float*) take((size_t)TT * II * 4);
    __bf16* h_bf  = (__bf16*)take((size_t)TT * HH * 2);
    __bf16* qb    = (__bf16*)take((size_t)TT * HQn  * DD * 2);
    __bf16* kb    = (__bf16*)take((size_t)TT * HKVn * DD * 2);
    __bf16* vb    = (__bf16*)take((size_t)TT * HKVn * DD * 2);
    __bf16* attnb = (__bf16*)take((size_t)TT * HH * 2);
    __bf16* actb  = (__bf16*)take((size_t)TT * II * 2);

    // 1. rmsnorm 1
    rmsnorm_bf16_kernel<<<TT, 256, 0, stream>>>(hidden, rms1_w, h_bf);

    // 2-4. Q / K / V projections
    gemm_bf16_wmma_kernel<0><<<dim3(HH / 128, TT / 128), 256, 0, stream>>>(
        h_bf, w_q, qf32, nullptr, nullptr, nullptr, HH, HH);
    gemm_bf16_wmma_kernel<0><<<dim3((HKVn * DD) / 128, TT / 128), 256, 0, stream>>>(
        h_bf, w_k, kf32, nullptr, nullptr, nullptr, HKVn * DD, HH);
    gemm_bf16_wmma_kernel<0><<<dim3((HKVn * DD) / 128, TT / 128), 256, 0, stream>>>(
        h_bf, w_v, vf32, nullptr, nullptr, nullptr, HKVn * DD, HH);

    // 5-7. RoPE (q, k) + V convert
    rope_bf16_kernel<<<(TT * HQn * 64) / 256, 256, 0, stream>>>(qf32, pos, qb, HQn);
    rope_bf16_kernel<<<(TT * HKVn * 64) / 256, 256, 0, stream>>>(kf32, pos, kb, HKVn);
    cvt_f32_bf16_kernel<<<(TT * HKVn * DD) / 256, 256, 0, stream>>>(vf32, vb, TT * HKVn * DD);

    // 8. flash attention
    flash_attn_kernel<<<dim3(TT / 64, HQn), 128, 0, stream>>>(qb, kb, vb, attnb);

    // 9. O projection + residual -> d_out
    gemm_bf16_wmma_kernel<1><<<dim3(HH / 128, TT / 128), 256, 0, stream>>>(
        attnb, w_o, out, nullptr, hidden, nullptr, HH, HH);

    // 10. rmsnorm 2
    rmsnorm_bf16_kernel<<<TT, 256, 0, stream>>>(out, rms2_w, h_bf);

    // 11. gate proj
    gemm_bf16_wmma_kernel<0><<<dim3(II / 128, TT / 128), 256, 0, stream>>>(
        h_bf, w_gate, gatef, nullptr, nullptr, nullptr, II, HH);

    // 12. up proj fused with silu(gate)*up -> bf16 act
    gemm_bf16_wmma_kernel<2><<<dim3(II / 128, TT / 128), 256, 0, stream>>>(
        h_bf, w_up, nullptr, actb, nullptr, gatef, II, HH);

    // 13. down proj + residual into d_out
    gemm_bf16_wmma_kernel<1><<<dim3(HH / 128, TT / 128), 256, 0, stream>>>(
        actb, w_down, out, nullptr, out, nullptr, HH, II);
}
